// mLSTM_8203387535515
// MI455X (gfx1250) — matrix operand order repaired and checked by reading, not compiled, not run on hardware
//
#include <hip/hip_runtime.h>
#include <hip/hip_bf16.h>
#include <math.h>

// ---------------------------------------------------------------------------
// mLSTM for MI455X (gfx1250): bf16 WMMA (V_WMMA_F32_16X16X32_BF16) for all
// GEMMs, persistent-kernel scan with device-atomic grid barriers.
// Sizes: B=16, S=512, I=512, H=1024, 4H=4096.
// ---------------------------------------------------------------------------

typedef __attribute__((ext_vector_type(16))) __bf16 v16bf;
typedef __attribute__((ext_vector_type(8)))  __bf16 v8bf;
typedef __attribute__((ext_vector_type(8)))  float  v8f;

#define B_SZ   16
#define S_SZ   512
#define I_SZ   512
#define H_SZ   1024
#define H4_SZ  4096
#define NBLK   64      // persistent workgroups in the scan kernel

// ---- WMMA tile operand loaders (wave32) -----------------------------------
// A: 16x32 bf16, row-major source, lda in elements.
// lanes 0-15: row=lane,   K = {k0+0..7,  k0+16..23}
// lanes16-31: row=lane-16,K = {k0+8..15, k0+24..31}
__device__ __forceinline__ v16bf load_A_tile(const __bf16* A, int lda, int k0, int lane) {
  const int half = lane >> 4;
  const int m    = lane & 15;
  const __bf16* p = A + (size_t)m * lda + k0 + half * 8;
  v8bf lo = *(const v8bf*)(p);
  v8bf hi = *(const v8bf*)(p + 16);
  v16bf a;
#pragma unroll
  for (int i = 0; i < 8; ++i) { a[i] = lo[i]; a[i + 8] = hi[i]; }
  return a;
}

// B: 32x16 bf16 = W^T tile; lane holds one column n = lane&15, which is one
// row of row-major W. lanes 0-15: K={k0+0..7, k0+16..23}, lanes 16-31: K={k0+8..15, k0+24..31}.
__device__ __forceinline__ v16bf load_B_tile(const __bf16* W, int ldb, int row0, int k0, int lane) {
  const int half = lane >> 4;
  const int n    = lane & 15;
  const __bf16* p = W + (size_t)(row0 + n) * ldb + k0 + half * 8;
  v8bf lo = *(const v8bf*)(p);
  v8bf hi = *(const v8bf*)(p + 16);
  v16bf b;
#pragma unroll
  for (int i = 0; i < 8; ++i) { b[i] = lo[i]; b[i + 8] = hi[i]; }
  return b;
}

// ---- Kernel 1: weight-norm rows, emit bf16 --------------------------------
__global__ __launch_bounds__(256) void normw_kernel(const float* __restrict__ v,
                                                    const float* __restrict__ g,
                                                    __bf16* __restrict__ out,
                                                    int rows, int cols) {
  const int wave = (blockIdx.x * blockDim.x + threadIdx.x) >> 5;
  const int lane = threadIdx.x & 31;
  if (wave >= rows) return;
  const float* vr = v + (size_t)wave * cols;
  float s = 0.f;
  for (int c = lane; c < cols; c += 32) { float t = vr[c]; s += t * t; }
#pragma unroll
  for (int off = 16; off > 0; off >>= 1) s += __shfl_down(s, off, 32);
  s = __shfl(s, 0, 32);
  const float scale = g[wave] * __frsqrt_rn(s);
  __bf16* orow = out + (size_t)wave * cols;
  for (int c = lane; c < cols; c += 32) orow[c] = (__bf16)(vr[c] * scale);
}

// ---- Kernel 2: cast x (B,S,I) f32 -> (S,B,I) bf16 -------------------------
__global__ __launch_bounds__(256) void castx_kernel(const float* __restrict__ x,
                                                    __bf16* __restrict__ xb) {
  const int idx = blockIdx.x * blockDim.x + threadIdx.x; // b*S*I + s*I + i
  const int i = idx & (I_SZ - 1);
  const int s = (idx >> 9) & (S_SZ - 1);
  const int b = idx >> 18;
  if (b >= B_SZ) return;
  xb[((size_t)(s * B_SZ + b) << 9) + i] = (__bf16)x[idx];
}

// ---- Kernel 3: mx = x@Wmx^T, zx = x@Wx^T (bf16 out, fp32 accum) -----------
// grid (80, 512), block 128 (4 waves). ntile 0..63 -> mx, 64..319 -> zx.
__global__ __launch_bounds__(128) void gemm_pre_kernel(const __bf16* __restrict__ xb,
                                                       const __bf16* __restrict__ wmx,
                                                       const __bf16* __restrict__ wx,
                                                       __bf16* __restrict__ mxb,
                                                       __bf16* __restrict__ zxb) {
  const int lane  = threadIdx.x & 31;
  const int wave  = threadIdx.x >> 5;
  const int ntile = blockIdx.x * 4 + wave;       // 0..319
  const int mtile = blockIdx.y;                  // 0..511
  const __bf16* A = xb + (size_t)mtile * 16 * I_SZ;
  const __bf16* W = (ntile < 64) ? wmx : wx;
  const int wrow  = (ntile < 64) ? ntile * 16 : (ntile - 64) * 16;

  v8f acc = {};
#pragma unroll 4
  for (int k0 = 0; k0 < I_SZ; k0 += 32) {
    v16bf a = load_A_tile(A, I_SZ, k0, lane);
    v16bf b = load_B_tile(W, I_SZ, wrow, k0, lane);
    acc = __builtin_amdgcn_wmma_f32_16x16x32_bf16(false, a, false, b, (short)0, acc, false, false);
  }
  const int half = lane >> 4, n = lane & 15;
#pragma unroll
  for (int v = 0; v < 8; ++v) {
    const int m   = v + half * 8;
    const int row = mtile * 16 + m;              // = s*16 + b
    if (ntile < 64) mxb[(size_t)row * H_SZ  + ntile * 16 + n]        = (__bf16)acc[v];
    else            zxb[(size_t)row * H4_SZ + (ntile - 64) * 16 + n] = (__bf16)acc[v];
  }
}

// ---- grid-wide barrier (monotonic counter, agent scope) -------------------
__device__ __forceinline__ void grid_barrier(unsigned* cnt, unsigned& k) {
  __syncthreads();
  if (threadIdx.x == 0) {
    __threadfence();
    __hip_atomic_fetch_add(cnt, 1u, __ATOMIC_RELEASE, __HIP_MEMORY_SCOPE_AGENT);
    ++k;
    const unsigned target = k * (unsigned)NBLK;
    while (__hip_atomic_load(cnt, __ATOMIC_ACQUIRE, __HIP_MEMORY_SCOPE_AGENT) < target)
      __builtin_amdgcn_s_sleep(1);
  }
  __syncthreads();
}

__device__ __forceinline__ float sigmoidf_(float x) { return 1.f / (1.f + __expf(-x)); }

// ---- Kernel 4: persistent recurrent scan ----------------------------------
// 64 WGs x 128 threads. WG g owns h-columns [g*16, g*16+16).
__global__ __launch_bounds__(128, 1) void mlstm_scan_kernel(
    const __bf16* __restrict__ wmh, const __bf16* __restrict__ wh,
    const __bf16* __restrict__ mxb, const __bf16* __restrict__ zxb,
    const float* __restrict__ whb,  const float* __restrict__ mask,
    float* __restrict__ h, float* __restrict__ c,
    __bf16* __restrict__ hb, __bf16* __restrict__ mb,
    float* __restrict__ out, unsigned* barrier_cnt) {
  __shared__ float lds[4 * 256];
  const int tid  = threadIdx.x;
  const int lane = tid & 31;
  const int wave = tid >> 5;
  const int g    = blockIdx.x;                   // h-tile index 0..63
  const int half = lane >> 4, nn = lane & 15;
  unsigned bk = 0;

  for (int t = 0; t < S_SZ; ++t) {
    // -- Phase A: partial (h @ Wmh^T) tile; 4 waves split K=1024 ------------
    {
      v8f acc = {};
      const int kbase = wave * 256;
#pragma unroll 2
      for (int kk = 0; kk < 256; kk += 32) {
        v16bf a = load_A_tile(hb, H_SZ, kbase + kk, lane);
        v16bf b = load_B_tile(wmh, H_SZ, g * 16, kbase + kk, lane);
        acc = __builtin_amdgcn_wmma_f32_16x16x32_bf16(false, a, false, b, (short)0, acc, false, false);
      }
#pragma unroll
      for (int v = 0; v < 8; ++v)
        lds[wave * 256 + (v + half * 8) * 16 + nn] = acc[v];
    }
    __syncthreads();
    // reduce K-partials, multiply by mx_t, emit m in bf16
#pragma unroll
    for (int r = 0; r < 2; ++r) {
      const int e  = tid + r * 128;              // 0..255 -> (m,n)
      const float s = lds[e] + lds[256 + e] + lds[512 + e] + lds[768 + e];
      const int m = e >> 4, n2 = e & 15;
      const int col = g * 16 + n2;
      const float mxv = (float)mxb[(size_t)(t * B_SZ + m) * H_SZ + col];
      mb[m * H_SZ + col] = (__bf16)(mxv * s);
    }
    grid_barrier(barrier_cnt, bk);

    // -- Phase B: z gate-block `wave` (i/f/o/u) for these 16 columns --------
    {
      v8f acc = {};
      const int zrow = wave * H_SZ + g * 16;     // Wh row base == z column base
#pragma unroll 2
      for (int k0 = 0; k0 < H_SZ; k0 += 32) {
        v16bf a = load_A_tile(mb, H_SZ, k0, lane);
        v16bf b = load_B_tile(wh, H_SZ, zrow, k0, lane);
        acc = __builtin_amdgcn_wmma_f32_16x16x32_bf16(false, a, false, b, (short)0, acc, false, false);
      }
#pragma unroll
      for (int v = 0; v < 8; ++v) {
        const int m  = v + half * 8;
        const int zc = zrow + nn;
        const float z = acc[v] + (float)zxb[(size_t)(t * B_SZ + m) * H4_SZ + zc] + whb[zc];
        lds[wave * 256 + (v + half * 8) * 16 + nn] = z;
      }
    }
    __syncthreads();
    // -- Gates + state update + hs store ------------------------------------
#pragma unroll
    for (int r = 0; r < 2; ++r) {
      const int e = tid + r * 128;
      const int m = e >> 4, n2 = e & 15;
      const int col = g * 16 + n2;
      const float i_ = sigmoidf_(lds[e]);
      const float f_ = sigmoidf_(lds[256 + e]);
      const float o_ = sigmoidf_(lds[512 + e]);
      const float u_ = tanhf(lds[768 + e]);
      const int hi = m * H_SZ + col;
      const float co = c[hi], ho = h[hi];
      float cn = f_ * co + i_ * u_;
      float hn = o_ * tanhf(cn);
      const float mt = mask[m * S_SZ + t];
      hn = mt * hn + (1.f - mt) * ho;
      cn = mt * cn + (1.f - mt) * co;
      h[hi] = hn; c[hi] = cn;
      hb[hi] = (__bf16)hn;
      out[((size_t)m * S_SZ + t) * H_SZ + col] = hn;           // hs (B,S,H)
      if (t == S_SZ - 1) {
        out[(size_t)B_SZ * S_SZ * H_SZ + hi]                 = hn;  // h_fin
        out[(size_t)B_SZ * S_SZ * H_SZ + B_SZ * H_SZ + hi]   = cn;  // c_fin
      }
    }
    grid_barrier(barrier_cnt, bk);
  }
}

// ---------------------------------------------------------------------------
extern "C" void kernel_launch(void* const* d_in, const int* in_sizes, int n_in,
                              void* d_out, int out_size, void* d_ws, size_t ws_size,
                              hipStream_t stream) {
  (void)in_sizes; (void)n_in; (void)out_size; (void)ws_size;
  const float* x     = (const float*)d_in[0];
  const float* mask  = (const float*)d_in[1];
  const float* wmx_v = (const float*)d_in[2];
  const float* wmx_g = (const float*)d_in[3];
  const float* wmh_v = (const float*)d_in[4];
  const float* wmh_g = (const float*)d_in[5];
  const float* wx_v  = (const float*)d_in[6];
  const float* wx_g  = (const float*)d_in[7];
  const float* wh_v  = (const float*)d_in[8];
  const float* wh_g  = (const float*)d_in[9];
  const float* wh_b  = (const float*)d_in[10];
  float* out = (float*)d_out;

  // workspace layout (all offsets 256-aligned)
  char* ws = (char*)d_ws;
  unsigned* bar   = (unsigned*)(ws + 0);                      // 256 B
  float*    h     = (float*)  (ws + 256);                     // 64 KB
  float*    c     = (float*)  (ws + 256 + 65536);             // 64 KB
  __bf16*   hb    = (__bf16*) (ws + 256 + 2 * 65536);         // 32 KB
  __bf16*   mb    = (__bf16*) (ws + 256 + 2 * 65536 + 32768); // 32 KB
  size_t off = 256 + 2 * 65536 + 2 * 32768;
  __bf16* xb   = (__bf16*)(ws + off); off += (size_t)B_SZ * S_SZ * I_SZ * 2;   // 8 MB
  __bf16* wmxb = (__bf16*)(ws + off); off += (size_t)H_SZ  * I_SZ * 2;         // 1 MB
  __bf16* wmhb = (__bf16*)(ws + off); off += (size_t)H_SZ  * H_SZ * 2;         // 2 MB
  __bf16* wxb  = (__bf16*)(ws + off); off += (size_t)H4_SZ * I_SZ * 2;         // 4 MB
  __bf16* whb_ = (__bf16*)(ws + off); off += (size_t)H4_SZ * H_SZ * 2;         // 8 MB
  __bf16* mxb  = (__bf16*)(ws + off); off += (size_t)S_SZ * B_SZ * H_SZ * 2;   // 16 MB
  __bf16* zxb  = (__bf16*)(ws + off); off += (size_t)S_SZ * B_SZ * H4_SZ * 2;  // 64 MB

  // zero barrier + h + c + hb every call (deterministic, capture-safe)
  hipMemsetAsync(ws, 0, 256 + 2 * 65536 + 32768, stream);

  // 1) weight normalization -> bf16
  normw_kernel<<<H_SZ  / 8, 256, 0, stream>>>(wmx_v, wmx_g, wmxb, H_SZ,  I_SZ);
  normw_kernel<<<H_SZ  / 8, 256, 0, stream>>>(wmh_v, wmh_g, wmhb, H_SZ,  H_SZ);
  normw_kernel<<<H4_SZ / 8, 256, 0, stream>>>(wx_v,  wx_g,  wxb,  H4_SZ, I_SZ);
  normw_kernel<<<H4_SZ / 8, 256, 0, stream>>>(wh_v,  wh_g,  whb_, H4_SZ, H_SZ);

  // 2) x -> (S,B,I) bf16
  castx_kernel<<<(B_SZ * S_SZ * I_SZ) / 256, 256, 0, stream>>>(x, xb);

  // 3) mx / zx precompute GEMMs
  gemm_pre_kernel<<<dim3(80, 512), 128, 0, stream>>>(xb, wmxb, wxb, mxb, zxb);

  // 4) persistent recurrent scan
  mlstm_scan_kernel<<<NBLK, 128, 0, stream>>>(wmhb, whb_, mxb, zxb, wh_b, mask,
                                              h, c, hb, mb, out, bar);
}
